// CGCN_36077725286436
// MI455X (gfx1250) — matrix operand order repaired and checked by reading, not compiled, not run on hardware
//
#include <hip/hip_runtime.h>

typedef __attribute__((ext_vector_type(16))) __bf16 v16bf;
typedef __attribute__((ext_vector_type(8)))  __bf16 v8bf;
typedef __attribute__((ext_vector_type(8)))  float  v8f;
typedef int v4i __attribute__((vector_size(16)));   // matches async-copy builtin pointee

#define BDIM   8
#define NNODE  2048
#define INDIM  256
#define HIDDIM 256
#define OUTDIM 128
#define MTOT   (BDIM * NNODE)   // 16384

#define LDT 40                   // LDS row stride in bf16 elems (32 + 8 pad, keeps 16B align)

// CDNA5 async global->LDS copy path (ASYNCcnt), guarded so either toolchain compiles
#if defined(__AMDGCN__) && __has_builtin(__builtin_amdgcn_global_load_async_to_lds_b128) && \
    __has_builtin(__builtin_amdgcn_s_wait_asynccnt)
#define USE_ASYNC_B 1
#else
#define USE_ASYNC_B 0
#endif

enum { EPI_DENSE = 0, EPI_PROP = 1 };

// ---------------------------------------------------------------------------
// dinv[b,n] = rsqrt(1 + sum_m adj[b,n,m]) ; one wave32 per row
// ---------------------------------------------------------------------------
__global__ __launch_bounds__(256)
void rowsum_dinv(const float* __restrict__ adj, float* __restrict__ dinv)
{
    const int lane = threadIdx.x & 31;
    const int wave = threadIdx.x >> 5;
    const long rb  = (long)blockIdx.x * 8 + wave;     // flat row in [0, B*N)
    const float4* row = (const float4*)(adj + rb * NNODE);
    float s = 0.0f;
    for (int i = lane; i < NNODE / 4; i += 32) {
        float4 v = row[i];
        s += v.x + v.y + v.z + v.w;
    }
#pragma unroll
    for (int off = 16; off > 0; off >>= 1) s += __shfl_xor(s, off, 32);
    if (lane == 0) dinv[rb] = rsqrtf(s + 1.0f);
}

// ---------------------------------------------------------------------------
// fp32 copy (output region 0) ; fp32 -> bf16 transposed weight convert
// ---------------------------------------------------------------------------
__global__ __launch_bounds__(256)
void copy_f32x4(const float4* __restrict__ src, float4* __restrict__ dst, int n4)
{
    int i = blockIdx.x * 256 + threadIdx.x;
    if (i < n4) dst[i] = src[i];
}

// dst[n*K + k] = (bf16) src[k*N + n]   (weights stored transposed for B staging)
__global__ __launch_bounds__(256)
void cvt_bf16_T(const float* __restrict__ src, __bf16* __restrict__ dst, int K, int N)
{
    int i = blockIdx.x * 256 + threadIdx.x;
    if (i < K * N) {
        int k = i / N, n = i % N;
        dst[(long)n * K + k] = (__bf16)src[i];
    }
}

// ---------------------------------------------------------------------------
// Tiled bf16 WMMA GEMM: C[M,Ncols] = act(A[M,K] @ B[K,Ncols])
//   A  : fp32 [M,K] row-major in global, converted to bf16 while staging to LDS
//   BT : bf16 [Ncols,K] (pre-transposed) in global -> straight b128 copies
//        (async global->LDS when the toolchain exposes the gfx1250 builtin)
//   Block: 256 thr = 8 waves (4 x 2), block tile 128x128, K-step 32,
//   double buffered, 1 barrier per K-step.
//   EPI_DENSE: v = relu(acc + bias[n]); Out[m,n]=v;
//              HAS_AUX: auxT[n,m] = bf16(dinv[m]*v)   (v8bf store, m-contiguous)
//   EPI_PROP : Out[m,n] = dinv[m] * (acc + (float)BT[n,m])  (identity-diag term)
// ---------------------------------------------------------------------------
template<int EPI, bool HAS_AUX>
__global__ __launch_bounds__(256)
void gemm_bf16_wmma(const float* __restrict__ Ag,
                    const __bf16* __restrict__ BTg,
                    float* __restrict__ Og,
                    const float* __restrict__ bias,
                    const float* __restrict__ dinvg,
                    __bf16* __restrict__ auxT,
                    int Kdim, int Ncols, long ldbT, long ldauxT,
                    long sA, long sBcol, long sO, long sD)
{
    const int z = blockIdx.z;
    const float*  A    = Ag    + (long)z * sA;
    const __bf16* BT   = BTg   + (long)z * sBcol;   // batch shifts along k axis
    float*        Out  = Og    + (long)z * sO;
    const float*  dinv = dinvg + (long)z * sD;

    const int tid  = threadIdx.x;
    const int lane = tid & 31;
    const int wave = tid >> 5;
    const int wm   = wave >> 1;           // 0..3
    const int wn   = wave & 1;            // 0..1
    const int bm   = blockIdx.y * 128;
    const int bn   = blockIdx.x * 128;

    __shared__ __bf16 lA[2][128 * LDT];   // [m][k]
    __shared__ __bf16 lB[2][128 * LDT];   // [n][k]

    v8f acc[2][4];
#pragma unroll
    for (int mt = 0; mt < 2; ++mt)
#pragma unroll
        for (int nt = 0; nt < 4; ++nt)
#pragma unroll
            for (int i = 0; i < 8; ++i) acc[mt][nt][i] = 0.0f;

    // ---- staging: each thread owns 16 contiguous elems of one A row + one BT row
    const int row = tid >> 1;             // 0..127
    const int cb  = (tid & 1) * 16;       // 0 | 16
    const float*  baseA = A  + (long)(bm + row) * Kdim + cb;
    const __bf16* baseB = BT + (long)(bn + row) * ldbT + cb;

    float4 ra[4];

    auto load_A = [&](int k0) {
        const float* srcA = baseA + k0;
#pragma unroll
        for (int q = 0; q < 4; ++q) ra[q] = ((const float4*)srcA)[q];
        if (k0 + 96 <= Kdim) __builtin_prefetch(srcA + 64, 0, 3);  // global_prefetch
    };
    auto store_A = [&](int buf) {
        __bf16* dA = &lA[buf][row * LDT + cb];
#pragma unroll
        for (int q = 0; q < 4; ++q) {
            dA[q * 4 + 0] = (__bf16)ra[q].x;
            dA[q * 4 + 1] = (__bf16)ra[q].y;
            dA[q * 4 + 2] = (__bf16)ra[q].z;
            dA[q * 4 + 3] = (__bf16)ra[q].w;
        }
    };

    // ---- fragment geometry (ISA 16-bit A 16x32 / B 32x16 lane layouts)
    const int arow  = wm * 32 + (lane & 15);
    const int koffA = (lane >> 4) * 8;
    const int brow  = wn * 64 + (lane & 15);
    const int koffB = (lane >> 4) * 8;

    auto compute = [&](int buf) {
        v16bf af[2], bfm[4];
#pragma unroll
        for (int mt = 0; mt < 2; ++mt) {
            const __bf16* p = &lA[buf][(arow + mt * 16) * LDT + koffA];
            v8bf lo = *(const v8bf*)p;            // K = off .. off+7
            v8bf hi = *(const v8bf*)(p + 16);     // K = off+16 .. off+23
#pragma unroll
            for (int j = 0; j < 8; ++j) { af[mt][j] = lo[j]; af[mt][8 + j] = hi[j]; }
        }
#pragma unroll
        for (int nt = 0; nt < 4; ++nt) {
            const __bf16* p = &lB[buf][(brow + nt * 16) * LDT + koffB];
            v8bf lo = *(const v8bf*)p;            // K = kb .. kb+7
            v8bf hi = *(const v8bf*)(p + 16);     // K = kb+16 .. kb+23
#pragma unroll
            for (int j = 0; j < 8; ++j) { bfm[nt][j] = lo[j]; bfm[nt][8 + j] = hi[j]; }
        }
#pragma unroll
        for (int mt = 0; mt < 2; ++mt)
#pragma unroll
            for (int nt = 0; nt < 4; ++nt)
                acc[mt][nt] = __builtin_amdgcn_wmma_f32_16x16x32_bf16(
                    false, af[mt], false, bfm[nt], (short)0, acc[mt][nt], false, false);
    };

#if USE_ASYNC_B
    // ---- async global->LDS for B (ASYNCcnt); issued post-barrier into the idle
    //      buffer so it overlaps a full compute phase, waited pre-barrier.
    auto stage_B = [&](int k0, int bufsel) {
        const __bf16* srcB = baseB + k0;
        __bf16* dB = &lB[bufsel][row * LDT + cb];
        __builtin_amdgcn_global_load_async_to_lds_b128(
            (__attribute__((address_space(1))) v4i*)srcB,
            (__attribute__((address_space(3))) v4i*)dB, 0, 0);
        __builtin_amdgcn_global_load_async_to_lds_b128(
            (__attribute__((address_space(1))) v4i*)(srcB + 8),
            (__attribute__((address_space(3))) v4i*)(dB + 8), 0, 0);
    };

    load_A(0);
    stage_B(0, 0);
    store_A(0);
    int buf = 0;
    for (int k0 = 32; k0 < Kdim; k0 += 32) {
        load_A(k0);                               // next A tile -> regs (LOADcnt)
        __builtin_amdgcn_s_wait_asynccnt(0);      // our lds[buf] B copies landed
        __syncthreads();                          // everyone's tile ready
        stage_B(k0, buf ^ 1);                     // async B for next tile
        compute(buf);
        store_A(buf ^ 1);
        buf ^= 1;
    }
    __builtin_amdgcn_s_wait_asynccnt(0);
    __syncthreads();
    compute(buf);
#else
    // ---- fallback: register-staged B double buffering ----
    v8bf rb0, rb1;
    auto load_B = [&](int k0) {
        const __bf16* srcB = baseB + k0;
        rb0 = *(const v8bf*)srcB;
        rb1 = *(const v8bf*)(srcB + 8);
    };
    auto store_B = [&](int buf) {
        __bf16* dB = &lB[buf][row * LDT + cb];
        *(v8bf*)dB       = rb0;
        *(v8bf*)(dB + 8) = rb1;
    };

    load_A(0); load_B(0);
    store_A(0); store_B(0);
    int buf = 0;
    for (int k0 = 32; k0 < Kdim; k0 += 32) {
        load_A(k0); load_B(k0);    // next tile's global loads in flight
        __syncthreads();           // lds[buf] ready for everyone
        compute(buf);
        store_A(buf ^ 1); store_B(buf ^ 1);
        buf ^= 1;
    }
    __syncthreads();
    compute(buf);
#endif

    // ---- epilogue; C layout: VGPR i -> M = (lane>>4)*8 + i, N = lane&15 ----
    const int mbase = bm + wm * 32 + (lane >> 4) * 8;
    const int nbase = bn + wn * 64 + (lane & 15);

    if constexpr (EPI == EPI_DENSE) {
        float bv[4];
#pragma unroll
        for (int nt = 0; nt < 4; ++nt) bv[nt] = bias[nbase + nt * 16];
        float dv[2][8];
        if constexpr (HAS_AUX) {
#pragma unroll
            for (int mt = 0; mt < 2; ++mt)
#pragma unroll
                for (int i = 0; i < 8; ++i) dv[mt][i] = dinv[mbase + mt * 16 + i];
        }
#pragma unroll
        for (int mt = 0; mt < 2; ++mt) {
#pragma unroll
            for (int nt = 0; nt < 4; ++nt) {
                const int n = nbase + nt * 16;
                v8bf avec;
#pragma unroll
                for (int i = 0; i < 8; ++i) {
                    const int m = mbase + mt * 16 + i;
                    float v = fmaxf(acc[mt][nt][i] + bv[nt], 0.0f);
                    Out[(long)m * Ncols + n] = v;
                    if constexpr (HAS_AUX) avec[i] = (__bf16)(dv[mt][i] * v);
                }
                if constexpr (HAS_AUX)
                    *(v8bf*)&auxT[(long)n * ldauxT + (mbase + mt * 16)] = avec;
            }
        }
    } else {
        float dv[2][8];
#pragma unroll
        for (int mt = 0; mt < 2; ++mt)
#pragma unroll
            for (int i = 0; i < 8; ++i) dv[mt][i] = dinv[mbase + mt * 16 + i];
#pragma unroll
        for (int mt = 0; mt < 2; ++mt) {
#pragma unroll
            for (int nt = 0; nt < 4; ++nt) {
                const int n = nbase + nt * 16;
                v8bf diag = *(const v8bf*)&BT[(long)n * ldbT + (mbase + mt * 16)];
#pragma unroll
                for (int i = 0; i < 8; ++i) {
                    const int m = mbase + mt * 16 + i;
                    Out[(long)m * Ncols + n] =
                        dv[mt][i] * (acc[mt][nt][i] + (float)diag[i]);
                }
            }
        }
    }
}

// ---------------------------------------------------------------------------
extern "C" void kernel_launch(void* const* d_in, const int* in_sizes, int n_in,
                              void* d_out, int out_size, void* d_ws, size_t ws_size,
                              hipStream_t stream)
{
    const float* adj   = (const float*)d_in[0];   // [B,N,N]
    const float* X     = (const float*)d_in[1];   // [B,N,IN]
    const float* wmean = (const float*)d_in[2];
    const float* bmean = (const float*)d_in[3];
    const float* w1    = (const float*)d_in[4];
    const float* b1    = (const float*)d_in[5];
    const float* w2    = (const float*)d_in[6];
    const float* b2    = (const float*)d_in[7];

    float* out      = (float*)d_out;
    float* out_x    = out;
    float* out_mean = out_x    + (long)MTOT * INDIM;
    float* out_hid  = out_mean + (long)MTOT * INDIM;
    float* out_out  = out_hid  + (long)MTOT * HIDDIM;

    char*   ws    = (char*)d_ws;
    float*  dinv  = (float*)(ws + 0);                          //  64 KB
    __bf16* wmbT  = (__bf16*)(ws + 65536);                     // 128 KB [256][256]
    __bf16* w1bT  = (__bf16*)(ws + 65536 + 131072);            // 128 KB [256][256]
    __bf16* w2bT  = (__bf16*)(ws + 65536 + 262144);            //  64 KB [128][256]
    __bf16* mvsT  = (__bf16*)(ws + 393216);                    //   8 MB [256][MTOT]
    __bf16* hidsT = (__bf16*)(ws + 393216 + 8388608);          //   8 MB [256][MTOT]
    float*  Ax    = (float*)(ws + 393216 + 16777216);          //  16 MB [MTOT][256]
    float*  Ah    = Ax;  // Ax dead after dense2 -> reuse

    dim3 blk(256);
    rowsum_dinv<<<MTOT / 8, blk, 0, stream>>>(adj, dinv);
    copy_f32x4<<<(MTOT * INDIM / 4) / 256, blk, 0, stream>>>(
        (const float4*)X, (float4*)out_x, MTOT * INDIM / 4);
    cvt_bf16_T<<<(INDIM * INDIM)   / 256, blk, 0, stream>>>(wmean, wmbT, INDIM,  INDIM);
    cvt_bf16_T<<<(INDIM * HIDDIM)  / 256, blk, 0, stream>>>(w1,    w1bT, INDIM,  HIDDIM);
    cvt_bf16_T<<<(HIDDIM * OUTDIM) / 256, blk, 0, stream>>>(w2,    w2bT, HIDDIM, OUTDIM);

    // mean = relu(X @ Wm + bm); mvsT[d, m] = bf16(dinv[m] * mean[m, d])
    gemm_bf16_wmma<EPI_DENSE, true><<<dim3(INDIM / 128, MTOT / 128, 1), blk, 0, stream>>>(
        X, wmbT, out_mean, bmean, dinv, mvsT,
        INDIM, INDIM, /*ldbT*/INDIM, /*ldauxT*/MTOT, 0, 0, 0, 0);

    // Ax = dinv_row * (adj @ mvs + mvs_diag)   (per batch; B k-axis shifts by NNODE)
    gemm_bf16_wmma<EPI_PROP, false><<<dim3(INDIM / 128, NNODE / 128, BDIM), blk, 0, stream>>>(
        adj, mvsT, Ax, nullptr, dinv, nullptr,
        NNODE, INDIM, /*ldbT*/MTOT, 0,
        (long)NNODE * NNODE, NNODE, (long)NNODE * INDIM, NNODE);

    // hid = relu(Ax @ W1 + b1); hidsT[d, m] = bf16(dinv[m] * hid[m, d])
    gemm_bf16_wmma<EPI_DENSE, true><<<dim3(HIDDIM / 128, MTOT / 128, 1), blk, 0, stream>>>(
        Ax, w1bT, out_hid, b1, dinv, hidsT,
        INDIM, HIDDIM, /*ldbT*/INDIM, /*ldauxT*/MTOT, 0, 0, 0, 0);

    // Ah = dinv_row * (adj @ hids + hids_diag)
    gemm_bf16_wmma<EPI_PROP, false><<<dim3(HIDDIM / 128, NNODE / 128, BDIM), blk, 0, stream>>>(
        adj, hidsT, Ah, nullptr, dinv, nullptr,
        NNODE, HIDDIM, /*ldbT*/MTOT, 0,
        (long)NNODE * NNODE, NNODE, (long)NNODE * HIDDIM, NNODE);

    // out = relu(Ah @ W2 + b2)
    gemm_bf16_wmma<EPI_DENSE, false><<<dim3(OUTDIM / 128, MTOT / 128, 1), blk, 0, stream>>>(
        Ah, w2bT, out_out, b2, dinv, nullptr,
        HIDDIM, OUTDIM, /*ldbT*/HIDDIM, /*ldauxT*/0, 0, 0, 0, 0);
}
